// nconv_gat_36292473651930
// MI455X (gfx1250) — matrix operand order repaired and checked by reading, not compiled, not run on hardware
//
#include <hip/hip_runtime.h>
#include <hip/hip_bf16.h>

// Problem constants (from reference setup_inputs)
#define BB 8
#define CC 64
#define TT 12
#define NN 512
#define EE 8192
#define EP (EE + NN)      // 8704 edges incl. self loops
#define GG (BB * TT)      // 96 graphs
#define HH 4
#define COC 64
#define GN (GG * NN)      // 49152 rows
#define HD (HH * COC)     // 256
#define NEG_SLOPE 0.2f

typedef __attribute__((ext_vector_type(16))) _Float16 v16h;
typedef __attribute__((ext_vector_type(8)))  _Float16 v8h;
typedef __attribute__((ext_vector_type(8)))  float    v8f;

// ---------- order-preserving float<->uint encoding for atomic max ----------
__device__ __forceinline__ unsigned encf(float x) {
    unsigned u = __float_as_uint(x);
    return (u & 0x80000000u) ? ~u : (u | 0x80000000u);
}
__device__ __forceinline__ float decf(unsigned u) {
    unsigned b = (u & 0x80000000u) ? (u ^ 0x80000000u) : ~u;
    return __uint_as_float(b);
}
__device__ __forceinline__ float lrelu(float x) {
    return x > 0.f ? x : NEG_SLOPE * x;
}

// ---------- prep: x[B,C,T,N] f32 -> xh[(b*T+t)*N+n][c] f16 ----------
__global__ void k_prep_x(const float* __restrict__ x, _Float16* __restrict__ xh) {
    int i = blockIdx.x * blockDim.x + threadIdx.x;
    if (i >= BB * CC * TT * NN) return;
    int n = i % NN;
    int t = (i / NN) % TT;
    int c = (i / (NN * TT)) % CC;
    int b = i / (NN * TT * CC);
    xh[((size_t)((b * TT + t) * NN + n)) * CC + c] = (_Float16)x[i];
}

// ---------- prep: W[64,256] f32 -> Wb[col][k] f16 (column major) ----------
__global__ void k_prep_w(const float* __restrict__ W, _Float16* __restrict__ Wb) {
    int i = blockIdx.x * blockDim.x + threadIdx.x;
    if (i >= CC * HD) return;
    int k = i / HD;
    int col = i % HD;
    Wb[col * CC + k] = (_Float16)W[i];
}

// ---------- init scratch: segment max/sum, degree + fill counters ----------
__global__ void k_init(unsigned* __restrict__ maxb, float* __restrict__ sumb,
                       int* __restrict__ deg, int* __restrict__ fill) {
    int i = blockIdx.x * blockDim.x + threadIdx.x;
    if (i < GN * HH) { maxb[i] = 0u; sumb[i] = 0.f; }
    if (i < NN)      { deg[i] = 0;  fill[i] = 0; }
}

// ---------- WMMA GEMM: h[GN,256] = xh[GN,64] * Wb^T ----------
// One wave per 16-row strip: A (16x64) loaded once into registers, then all
// 16 column tiles computed with 2 chained 16x16x32 f16 WMMAs each (32 WMMAs
// per wave).  16x less A traffic than one-wave-per-tile.
__global__ void __launch_bounds__(256) k_gemm(const _Float16* __restrict__ xh,
                                              const _Float16* __restrict__ Wb,
                                              float* __restrict__ hbuf) {
    int lane = threadIdx.x & 31;
    int wave = threadIdx.x >> 5;
    int row_tile = blockIdx.x * 8 + wave;   // 3072 row tiles
    int half = lane >> 4;                   // lane group 0-15 / 16-31
    int m    = lane & 15;                   // A row within tile / B column within tile

    const _Float16* rowA = xh + (size_t)(row_tile * 16 + m) * CC;

    // A 16x32 f16 layout: half h -> elems[0..7]=K ka+8h.., elems[8..15]=K ka+16+8h..
    v16h A0, A1;
    {
        const v8h a00 = *(const v8h*)(rowA + half * 8);
        const v8h a01 = *(const v8h*)(rowA + 16 + half * 8);
        const v8h a10 = *(const v8h*)(rowA + 32 + half * 8);
        const v8h a11 = *(const v8h*)(rowA + 48 + half * 8);
#pragma unroll
        for (int i = 0; i < 8; i++) {
            A0[i] = a00[i]; A0[i + 8] = a01[i];
            A1[i] = a10[i]; A1[i + 8] = a11[i];
        }
    }

    float* outbase = hbuf + (size_t)(row_tile * 16) * HD + m;
#pragma unroll
    for (int ct = 0; ct < HD / 16; ct++) {
        const _Float16* colB = Wb + (size_t)(ct * 16 + m) * CC;
        // B 32x16 f16 layout (same as A): half h -> elems[0..7]=K ka+8h.., elems[8..15]=K ka+16+8h..
        const v8h b00 = *(const v8h*)(colB + half * 8);
        const v8h b01 = *(const v8h*)(colB + 16 + half * 8);
        const v8h b10 = *(const v8h*)(colB + 32 + half * 8);
        const v8h b11 = *(const v8h*)(colB + 48 + half * 8);
        v16h B0, B1;
#pragma unroll
        for (int i = 0; i < 8; i++) {
            B0[i] = b00[i]; B0[i + 8] = b01[i];
            B1[i] = b10[i]; B1[i + 8] = b11[i];
        }
        v8f acc = {};
        acc = __builtin_amdgcn_wmma_f32_16x16x32_f16(
            false, A0, false, B0, (short)0, acc, false, false);
        acc = __builtin_amdgcn_wmma_f32_16x16x32_f16(
            false, A1, false, B1, (short)0, acc, false, false);
        // D layout: VGPR r, lanes0-15 -> M=r, lanes16-31 -> M=r+8; N = lane&15
        float* outp = outbase + ct * 16;
#pragma unroll
        for (int r = 0; r < 8; r++) {
            int M = r + 8 * half;
            outp[(size_t)M * HD] = acc[r];
        }
    }
}

// ---------- attention dots: a_src/a_dst[gn*H+h] ----------
__global__ void k_att(const float* __restrict__ hbuf, const float* __restrict__ att_src,
                      const float* __restrict__ att_dst, float* __restrict__ asrc,
                      float* __restrict__ adst) {
    int t = blockIdx.x * blockDim.x + threadIdx.x;
    if (t >= GN * HH) return;
    int head = t & 3;
    int gn   = t >> 2;
    const float4* hp = (const float4*)(hbuf + (size_t)gn * HD + head * COC);
    const float4* ps = (const float4*)(att_src + head * COC);
    const float4* pd = (const float4*)(att_dst + head * COC);
    float s = 0.f, d = 0.f;
#pragma unroll
    for (int i = 0; i < 16; i++) {
        float4 hv = hp[i], sv = ps[i], dv = pd[i];
        s += hv.x * sv.x + hv.y * sv.y + hv.z * sv.z + hv.w * sv.w;
        d += hv.x * dv.x + hv.y * dv.y + hv.z * dv.z + hv.w * dv.w;
    }
    asrc[t] = s;
    adst[t] = d;
}

// ---------- segment max over edges (order-preserving uint atomicMax) ----------
__global__ void k_edge_max(const int* __restrict__ ei, const float* __restrict__ asrc,
                           const float* __restrict__ adst, unsigned* __restrict__ maxb) {
    int t = blockIdx.x * blockDim.x + threadIdx.x;
    if (t >= EP * GG * HH) return;
    int h = t & 3;
    int g = (t >> 2) % GG;
    int e = (t >> 2) / GG;
    int src, dst;
    if (e < EE) { src = ei[e]; dst = ei[EE + e]; } else { src = dst = e - EE; }
    float x = lrelu(asrc[(g * NN + src) * HH + h] + adst[(g * NN + dst) * HH + h]);
    atomicMax(&maxb[(g * NN + dst) * HH + h], encf(x));
}

// ---------- segment sum of exp(logit - max) ----------
__global__ void k_edge_sum(const int* __restrict__ ei, const float* __restrict__ asrc,
                           const float* __restrict__ adst, const unsigned* __restrict__ maxb,
                           float* __restrict__ sumb) {
    int t = blockIdx.x * blockDim.x + threadIdx.x;
    if (t >= EP * GG * HH) return;
    int h = t & 3;
    int g = (t >> 2) % GG;
    int e = (t >> 2) / GG;
    int src, dst;
    if (e < EE) { src = ei[e]; dst = ei[EE + e]; } else { src = dst = e - EE; }
    int di = (g * NN + dst) * HH + h;
    float x = lrelu(asrc[(g * NN + src) * HH + h] + adst[di]);
    atomicAdd(&sumb[di], __expf(x - decf(maxb[di])));
}

// ---------- CSR build: degree count ----------
__global__ void k_deg(const int* __restrict__ ei, int* __restrict__ deg) {
    int e = blockIdx.x * blockDim.x + threadIdx.x;
    if (e >= EP) return;
    int dst = (e < EE) ? ei[EE + e] : (e - EE);
    atomicAdd(&deg[dst], 1);
}

// ---------- CSR build: exclusive prefix sum (single block, N=512) ----------
__global__ void k_scan(const int* __restrict__ deg, int* __restrict__ offs) {
    __shared__ int sh[NN];
    int tid = threadIdx.x;
    sh[tid] = deg[tid];
    __syncthreads();
    for (int o = 1; o < NN; o <<= 1) {
        int v = (tid >= o) ? sh[tid - o] : 0;
        __syncthreads();
        sh[tid] += v;
        __syncthreads();
    }
    offs[tid + 1] = sh[tid];
    if (tid == 0) offs[0] = 0;
}

// ---------- CSR build: fill edge ids by destination ----------
__global__ void k_fill(const int* __restrict__ ei, const int* __restrict__ offs,
                       int* __restrict__ fill, int* __restrict__ csr) {
    int e = blockIdx.x * blockDim.x + threadIdx.x;
    if (e >= EP) return;
    int dst = (e < EE) ? ei[EE + e] : (e - EE);
    int pos = atomicAdd(&fill[dst], 1);
    csr[offs[dst] + pos] = e;
}

// ---------- aggregation: one wave per (g, n); lanes cover 64 out channels ----------
__global__ void __launch_bounds__(256) k_agg(const int* __restrict__ ei, const int* __restrict__ offs,
                                             const int* __restrict__ csr, const float* __restrict__ hbuf,
                                             const float* __restrict__ asrc, const float* __restrict__ adst,
                                             const unsigned* __restrict__ maxb, const float* __restrict__ sumb,
                                             const float* __restrict__ bias, float* __restrict__ out) {
    int lane = threadIdx.x & 31;
    int wave = threadIdx.x >> 5;
    int gw   = blockIdx.x * 8 + wave;
    int g = gw / NN;
    int n = gw % NN;

    float ad[HH], mx[HH], inv_s[HH];
#pragma unroll
    for (int h = 0; h < HH; h++) {
        int di = (g * NN + n) * HH + h;
        ad[h]    = adst[di];
        mx[h]    = decf(maxb[di]);
        inv_s[h] = 1.0f / sumb[di];
    }
    float acc0 = 0.f, acc1 = 0.f;
    int beg = offs[n], end = offs[n + 1];
    for (int k = beg; k < end; k++) {
        int e = csr[k];
        int src = (e < EE) ? ei[e] : (e - EE);
        const float* hrow = hbuf + (size_t)(g * NN + src) * HD;
        const float* asp  = asrc + (g * NN + src) * HH;
#pragma unroll
        for (int h = 0; h < HH; h++) {
            float x = lrelu(asp[h] + ad[h]);
            float w = __expf(x - mx[h]) * inv_s[h];
            acc0 += w * hrow[h * COC + lane];
            acc1 += w * hrow[h * COC + lane + 32];
        }
    }
    int b = g / TT, t = g % TT;
    const float inv_h = 1.0f / HH;
    out[((size_t)(b * COC + lane)      * TT + t) * NN + n] = acc0 * inv_h + bias[lane];
    out[((size_t)(b * COC + lane + 32) * TT + t) * NN + n] = acc1 * inv_h + bias[lane + 32];
}

extern "C" void kernel_launch(void* const* d_in, const int* in_sizes, int n_in,
                              void* d_out, int out_size, void* d_ws, size_t ws_size,
                              hipStream_t stream) {
    (void)in_sizes; (void)n_in; (void)out_size; (void)ws_size;
    const float* x        = (const float*)d_in[0];
    const int*   ei       = (const int*)d_in[1];
    const float* W        = (const float*)d_in[2];
    const float* att_src  = (const float*)d_in[3];
    const float* att_dst  = (const float*)d_in[4];
    const float* bias     = (const float*)d_in[5];
    float*       out      = (float*)d_out;

    // workspace carve-out (256B aligned)
    size_t off = 0;
    char* wsb = (char*)d_ws;
    auto alloc = [&](size_t bytes) -> void* {
        void* p = wsb + off;
        off += (bytes + 255) & ~(size_t)255;
        return p;
    };
    _Float16* xh   = (_Float16*)alloc(sizeof(_Float16) * (size_t)GN * CC);  // 6.3 MB
    _Float16* Wb   = (_Float16*)alloc(sizeof(_Float16) * CC * HD);          // 32 KB
    float*    hbuf = (float*)   alloc(sizeof(float) * (size_t)GN * HD);     // 50.3 MB
    float*    asrc = (float*)   alloc(sizeof(float) * (size_t)GN * HH);
    float*    adst = (float*)   alloc(sizeof(float) * (size_t)GN * HH);
    unsigned* maxb = (unsigned*)alloc(sizeof(unsigned) * (size_t)GN * HH);
    float*    sumb = (float*)   alloc(sizeof(float) * (size_t)GN * HH);
    int*      deg  = (int*)     alloc(sizeof(int) * NN);
    int*      fill = (int*)     alloc(sizeof(int) * NN);
    int*      offs = (int*)     alloc(sizeof(int) * (NN + 1));
    int*      csr  = (int*)     alloc(sizeof(int) * EP);

    const int thr = 256;
    k_prep_x<<<(BB * CC * TT * NN + thr - 1) / thr, thr, 0, stream>>>(x, xh);
    k_prep_w<<<(CC * HD + thr - 1) / thr, thr, 0, stream>>>(W, Wb);
    k_init<<<(GN * HH + thr - 1) / thr, thr, 0, stream>>>(maxb, sumb, deg, fill);

    // 3072 row-strip waves; 8 waves/block -> 384 blocks; 32 WMMAs per wave
    k_gemm<<<(GN / 16) / 8, 256, 0, stream>>>(xh, Wb, hbuf);

    k_att<<<(GN * HH + thr - 1) / thr, thr, 0, stream>>>(hbuf, att_src, att_dst, asrc, adst);

    int edge_threads = EP * GG * HH;
    k_edge_max<<<(edge_threads + thr - 1) / thr, thr, 0, stream>>>(ei, asrc, adst, maxb);
    k_edge_sum<<<(edge_threads + thr - 1) / thr, thr, 0, stream>>>(ei, asrc, adst, maxb, sumb);

    k_deg<<<(EP + thr - 1) / thr, thr, 0, stream>>>(ei, deg);
    k_scan<<<1, NN, 0, stream>>>(deg, offs);
    k_fill<<<(EP + thr - 1) / thr, thr, 0, stream>>>(ei, offs, fill, csr);

    // 96*512 = 49152 waves; 8 waves/block
    k_agg<<<GG * NN / 8, 256, 0, stream>>>(ei, offs, csr, hbuf, asrc, adst, maxb, sumb, bias, out);
}
